// LSTMTaggerOwn_48996986912981
// MI455X (gfx1250) — matrix operand order repaired and checked by reading, not compiled, not run on hardware
//
#include <hip/hip_runtime.h>
#include <hip/hip_bf16.h>

// ---------------------------------------------------------------------------
// Problem constants (from reference)
// ---------------------------------------------------------------------------
static constexpr int SEQLEN = 1024;
static constexpr int EMB    = 2048;
static constexpr int HID    = 2048;
static constexpr int G4     = 4 * HID;   // 8192
static constexpr int NTAGS  = 50;
static constexpr int NWG_SCAN = 128;     // 2048 h-elements / 16 per workgroup

// CDNA5 WMMA vector types
typedef __attribute__((ext_vector_type(16))) __bf16 v16bf;
typedef __attribute__((ext_vector_type(8)))  float  v8f;

// ---------------------------------------------------------------------------
// Helpers
// ---------------------------------------------------------------------------
__device__ __forceinline__ unsigned short f2bf(float f) {
    unsigned int u = __float_as_uint(f);
    u += 0x7FFFu + ((u >> 16) & 1u);          // round-to-nearest-even
    return (unsigned short)(u >> 16);
}
__device__ __forceinline__ float bf2f(unsigned int lo16) {
    return __uint_as_float(lo16 << 16);
}

// Device-scope generation barrier (one caller thread per workgroup).
__device__ __forceinline__ void grid_barrier(int* cnt, int* gen, int nwg) {
    int g = __hip_atomic_load(gen, __ATOMIC_ACQUIRE, __HIP_MEMORY_SCOPE_AGENT);
    int a = __hip_atomic_fetch_add(cnt, 1, __ATOMIC_ACQ_REL, __HIP_MEMORY_SCOPE_AGENT);
    if (a == nwg - 1) {
        __hip_atomic_store(cnt, 0, __ATOMIC_RELAXED, __HIP_MEMORY_SCOPE_AGENT);
        __hip_atomic_fetch_add(gen, 1, __ATOMIC_RELEASE, __HIP_MEMORY_SCOPE_AGENT);
    } else {
        while (__hip_atomic_load(gen, __ATOMIC_ACQUIRE, __HIP_MEMORY_SCOPE_AGENT) == g) {
            __builtin_amdgcn_s_sleep(8);
        }
    }
}

// ---------------------------------------------------------------------------
// Kernel 0: bias = b_ih + b_hh, zero h buffer, zero barrier state
// ---------------------------------------------------------------------------
__global__ __launch_bounds__(256) void prep_kernel(
    const float* __restrict__ b_ih, const float* __restrict__ b_hh,
    float* __restrict__ bias, float* __restrict__ hbuf, int* __restrict__ bar)
{
    int i = blockIdx.x * 256 + threadIdx.x;
    if (i < G4)  bias[i] = b_ih[i] + b_hh[i];
    if (i < HID) hbuf[i] = 0.0f;
    if (i < 2)   bar[i]  = 0;
}

// ---------------------------------------------------------------------------
// Kernel 1a: gather embeddings for the sequence and convert to bf16
//            xs[t][k] = bf16(emb[seq[t]][k])
// ---------------------------------------------------------------------------
__global__ __launch_bounds__(256) void gather_kernel(
    const int* __restrict__ seq, const float* __restrict__ emb,
    unsigned short* __restrict__ xs)
{
    int idx = blockIdx.x * 256 + threadIdx.x;        // < SEQLEN*EMB
    int row = idx >> 11;                             // /EMB
    int k   = idx & (EMB - 1);
    xs[idx] = f2bf(emb[(size_t)seq[row] * EMB + k]);
}

// ---------------------------------------------------------------------------
// Kernel 1b: convert W_ih (f32) -> bf16 once, vectorized (4 elems / thread)
// ---------------------------------------------------------------------------
__global__ __launch_bounds__(256) void cvt_wih_kernel(
    const float* __restrict__ W, unsigned short* __restrict__ Wb)
{
    size_t i4 = ((size_t)blockIdx.x * 256 + threadIdx.x) * 4;  // < 8192*2048
    float4 f = *(const float4*)(W + i4);
    uint2 packed;
    packed.x = (unsigned)f2bf(f.x) | ((unsigned)f2bf(f.y) << 16);
    packed.y = (unsigned)f2bf(f.z) | ((unsigned)f2bf(f.w) << 16);
    *(uint2*)(Wb + i4) = packed;
}

// ---------------------------------------------------------------------------
// Kernel 2: x_gates[1024][8192] = xs @ W_ih^T + bias   (bf16 WMMA, f32 accum)
// Each wave: 64x32 output block (4 M-tiles x 2 N-tiles) -> 8 WMMAs share
// 12 b128 loads per K-iteration. Pure load+WMMA inner loop (no cvt VALU).
// ---------------------------------------------------------------------------
__global__ __launch_bounds__(256) void gemm_xgates_kernel(
    const unsigned short* __restrict__ xs,   // [1024][2048] bf16
    const unsigned short* __restrict__ Wb,   // [8192][2048] bf16
    const float* __restrict__ bias,          // [8192]
    float* __restrict__ xg)                  // [1024][8192]
{
    const int lane = threadIdx.x & 31;
    const int wave = threadIdx.x >> 5;
    const int gw   = blockIdx.x * 8 + wave;  // 4096 waves total
    const int n_pair  = gw & 255;            // 8192/32 = 256 N pairs
    const int m_strip = gw >> 8;             // 1024/64 = 16 M strips
    const int n_base  = n_pair * 32;
    const int m_base  = m_strip * 64;

    // B lane mapping: lane&15 = column within tile; lanes 16-31 take K+16
    const int col0 = n_base + (lane & 15);          // first N tile column
    const int col1 = col0 + 16;                     // second N tile column
    const int kShB = (lane < 16) ? 0 : 16;          // B: K 0-15 vs 16-31
    // A lane mapping: lane&15 = row; lanes 0-15 K 0-15, lanes 16-31 K 16-31
    const int kShA = (lane < 16) ? 0 : 16;
    const int arow = lane & 15;

    v8f acc[4][2];
#pragma unroll
    for (int s = 0; s < 4; ++s) { acc[s][0] = v8f{}; acc[s][1] = v8f{}; }

    const unsigned short* b0p = Wb + (size_t)col0 * EMB + kShB;
    const unsigned short* b1p = Wb + (size_t)col1 * EMB + kShB;

    for (int k0 = 0; k0 < EMB; k0 += 32) {
        // ---- two B tiles: 16 consecutive bf16 K values per lane (2 x b128)
        union { uint4 q[2]; v16bf v; } B0, B1;
        B0.q[0] = ((const uint4*)(b0p + k0))[0];
        B0.q[1] = ((const uint4*)(b0p + k0))[1];
        B1.q[0] = ((const uint4*)(b1p + k0))[0];
        B1.q[1] = ((const uint4*)(b1p + k0))[1];

        // ---- 4 A tiles, each feeds 2 WMMAs
#pragma unroll
        for (int s = 0; s < 4; ++s) {
            const uint4* ap = (const uint4*)
                (xs + (size_t)(m_base + s * 16 + arow) * EMB + k0 + kShA);
            union { uint4 q[2]; v16bf v; } A;
            A.q[0] = ap[0];      // K offsets +0..7   (8 bf16 = 16 B)
            A.q[1] = ap[1];      // K offsets +8..15
            acc[s][0] = __builtin_amdgcn_wmma_f32_16x16x32_bf16(
                false, A.v, false, B0.v, (short)0, acc[s][0], false, false);
            acc[s][1] = __builtin_amdgcn_wmma_f32_16x16x32_bf16(
                false, A.v, false, B1.v, (short)0, acc[s][1], false, false);
        }
    }

    // ---- epilogue: D(m,n): lane<16 -> M=v, lane>=16 -> M=v+8 ; N = lane&15
    const float bv0 = bias[col0];
    const float bv1 = bias[col1];
    const int rofs = (lane & 16) ? 8 : 0;
#pragma unroll
    for (int s = 0; s < 4; ++s) {
#pragma unroll
        for (int v = 0; v < 8; ++v) {
            int row = m_base + s * 16 + v + rofs;
            xg[(size_t)row * G4 + col0] = acc[s][0][v] + bv0;
            xg[(size_t)row * G4 + col1] = acc[s][1][v] + bv1;
        }
    }
}

// ---------------------------------------------------------------------------
// Kernel 3: persistent LSTM scan. 128 workgroups; WG `wg` owns h[16*wg..+15]
// and keeps its 64 W_hh rows (bf16) resident in LDS (256 KB).
// ---------------------------------------------------------------------------
__global__ __launch_bounds__(256) void lstm_scan_kernel(
    const float* __restrict__ W_hh,    // [8192][2048]
    const float* __restrict__ xg,      // [1024][8192] pre-activations (+bias)
    float* __restrict__ hbuf,          // [2048] global h (init 0)
    int* __restrict__ bar)             // barrier {cnt, gen}
{
    extern __shared__ char smem[];
    unsigned short* w_lds = (unsigned short*)smem;              // 64*2048 bf16
    float* h_lds = (float*)(smem + (size_t)64 * 2048 * 2);      // 2048 f32
    float* part  = h_lds + 2048;                                // 256
    float* gate  = part + 256;                                  // 64
    float* c_lds = gate + 64;                                   // 16

    const int tid = threadIdx.x;
    const int wg  = blockIdx.x;

    // Load + convert this WG's 64 W_hh rows into LDS (once).
    for (int idx = tid; idx < 64 * 2048; idx += 256) {
        int r = idx >> 11;               // local row 0..63
        int k = idx & 2047;
        int grow = (r >> 4) * HID + wg * 16 + (r & 15);   // quadrant*2048 + slice
        w_lds[idx] = f2bf(W_hh[(size_t)grow * HID + k]);
    }
    if (tid < 16) c_lds[tid] = 0.0f;
    __syncthreads();

    const int r = tid >> 2;              // local gate row 0..63
    const int p = tid & 3;               // K quarter
    const unsigned short* wr = w_lds + r * 2048 + p * 512;
    const float* hp = h_lds + p * 512;

    for (int t = 0; t < SEQLEN; ++t) {
        // pull full h (previous step) into LDS
        for (int i = tid; i < HID; i += 256) h_lds[i] = hbuf[i];
        __syncthreads();

        // partial dot: 512 MACs per thread, 4 threads per gate row
        float acc = 0.0f;
#pragma unroll 4
        for (int k = 0; k < 512; k += 4) {
            uint2  wv = *(const uint2*)(wr + k);
            float4 hv = *(const float4*)(hp + k);
            acc = fmaf(hv.x, bf2f(wv.x & 0xFFFFu), acc);
            acc = fmaf(hv.y, bf2f(wv.x >> 16), acc);
            acc = fmaf(hv.z, bf2f(wv.y & 0xFFFFu), acc);
            acc = fmaf(hv.w, bf2f(wv.y >> 16), acc);
        }
        part[tid] = acc;
        __syncthreads();

        if (tid < 64)
            gate[tid] = part[tid * 4] + part[tid * 4 + 1]
                      + part[tid * 4 + 2] + part[tid * 4 + 3];
        __syncthreads();

        if (tid < 16) {
            const int col = wg * 16 + tid;
            const float* x = xg + (size_t)t * G4;
            float gi = gate[tid]      + x[col];
            float gf = gate[16 + tid] + x[HID + col];
            float gg = gate[32 + tid] + x[2 * HID + col];
            float go = gate[48 + tid] + x[3 * HID + col];
            float iv = 1.0f / (1.0f + __expf(-gi));
            float fv = 1.0f / (1.0f + __expf(-gf));
            float gv = tanhf(gg);
            float ov = 1.0f / (1.0f + __expf(-go));
            float c  = fv * c_lds[tid] + iv * gv;
            c_lds[tid] = c;
            hbuf[col] = ov * tanhf(c);
        }
        __threadfence();
        __syncthreads();
        if (tid == 0) grid_barrier(bar, bar + 1, (int)gridDim.x);
        __syncthreads();
    }
}

// ---------------------------------------------------------------------------
// Kernel 4: tag head + log_softmax (tiny)
// ---------------------------------------------------------------------------
__global__ __launch_bounds__(64) void tag_head_kernel(
    const float* __restrict__ hbuf, const float* __restrict__ W_tag,
    const float* __restrict__ b_tag, float* __restrict__ out)
{
    __shared__ float s[64];
    __shared__ float lse;
    const int tid = threadIdx.x;
    if (tid < NTAGS) {
        float acc = b_tag[tid];
        const float* w = W_tag + (size_t)tid * HID;
        for (int k = 0; k < HID; k += 4) {
            float4 hv = *(const float4*)(hbuf + k);
            float4 wv = *(const float4*)(w + k);
            acc = fmaf(hv.x, wv.x, acc);
            acc = fmaf(hv.y, wv.y, acc);
            acc = fmaf(hv.z, wv.z, acc);
            acc = fmaf(hv.w, wv.w, acc);
        }
        s[tid] = acc;
    }
    __syncthreads();
    if (tid == 0) {
        float m = s[0];
        for (int i = 1; i < NTAGS; ++i) m = fmaxf(m, s[i]);
        float sum = 0.0f;
        for (int i = 0; i < NTAGS; ++i) sum += __expf(s[i] - m);
        lse = m + __logf(sum);
    }
    __syncthreads();
    if (tid < NTAGS) out[tid] = s[tid] - lse;
}

// ---------------------------------------------------------------------------
// Launch
// ---------------------------------------------------------------------------
extern "C" void kernel_launch(void* const* d_in, const int* in_sizes, int n_in,
                              void* d_out, int out_size, void* d_ws, size_t ws_size,
                              hipStream_t stream) {
    const int*   seq   = (const int*)d_in[0];
    const float* emb   = (const float*)d_in[1];
    const float* W_ih  = (const float*)d_in[2];
    const float* W_hh  = (const float*)d_in[3];
    const float* b_ih  = (const float*)d_in[4];
    const float* b_hh  = (const float*)d_in[5];
    const float* W_tag = (const float*)d_in[6];
    const float* b_tag = (const float*)d_in[7];
    float* out = (float*)d_out;

    // Workspace layout
    char* ws = (char*)d_ws;
    unsigned short* xs   = (unsigned short*)ws;                      //  4 MB bf16 xs
    float*          xg   = (float*)(ws + (size_t)4  * 1024 * 1024);  // 32 MB x_gates
    unsigned short* Wihb = (unsigned short*)(ws + (size_t)36 * 1024 * 1024); // 32 MB
    float*          bias = (float*)(ws + (size_t)68 * 1024 * 1024);  // 32 KB
    float*          hbuf = bias + G4;                                //  8 KB
    int*            bar  = (int*)(hbuf + HID);                       //  8 B

    prep_kernel<<<32, 256, 0, stream>>>(b_ih, b_hh, bias, hbuf, bar);
    gather_kernel<<<(SEQLEN * EMB) / 256, 256, 0, stream>>>(seq, emb, xs);
    cvt_wih_kernel<<<(G4 * EMB) / (256 * 4), 256, 0, stream>>>(W_ih, Wihb);
    gemm_xgates_kernel<<<512, 256, 0, stream>>>(xs, Wihb, bias, xg);

    size_t lds = (size_t)64 * 2048 * 2 + (2048 + 256 + 64 + 16) * sizeof(float);
    lstm_scan_kernel<<<NWG_SCAN, 256, lds, stream>>>(W_hh, xg, hbuf, bar);

    tag_head_kernel<<<1, 64, 0, stream>>>(hbuf, W_tag, b_tag, out);
}
